// FirstLayerAggregator_53429393162862
// MI455X (gfx1250) — matrix operand order repaired and checked by reading, not compiled, not run on hardware
//
#include <hip/hip_runtime.h>
#include <hip/hip_bf16.h>
#include <math.h>

typedef __attribute__((ext_vector_type(16))) _Float16 v16h;
typedef __attribute__((ext_vector_type(8)))  float    v8f;
typedef __attribute__((ext_vector_type(4)))  _Float16 v4h;

#define IN_DIM  64
#define OUT_DIM 32
#define ALPHA   0.2f

// Native fp32 atomic add: relaxed + agent scope so the backend selects
// global_atomic_add_f32 (L2 atomic units) instead of a CAS loop.
__device__ __forceinline__ void atomic_add_f32(float* p, float v) {
    __hip_atomic_fetch_add(p, v, __ATOMIC_RELAXED, __HIP_MEMORY_SCOPE_AGENT);
}

// ---------------- zero accumulators ----------------
__global__ void k_zero(float* __restrict__ p, long long n) {
    long long i = (long long)blockIdx.x * blockDim.x + threadIdx.x;
    if (i < n) p[i] = 0.f;
}

// ---------------- gather + dual GEMM (WMMA f16 -> f32) ----------------
// One wave computes a 16x32 tile of h for BOTH branches (x read once).
__global__ __launch_bounds__(256)
void k_gemm(const float* __restrict__ emb,
            const int*   __restrict__ uids,
            const float* __restrict__ Wb,
            const float* __restrict__ Wu,
            _Float16* __restrict__ hb,
            _Float16* __restrict__ hu,
            int M) {
    // weights transposed in LDS: sW[br][n*64 + k] = W[k][n]  (f16)
    __shared__ _Float16 sW[2][IN_DIM * OUT_DIM];
    __shared__ _Float16 sT[8][16 * OUT_DIM];   // per-wave 16x32 output tile

    for (int t = threadIdx.x; t < IN_DIM * OUT_DIM; t += 256) {
        int k = t >> 5, n = t & 31;
        sW[0][n * IN_DIM + k] = (_Float16)Wb[t];
        sW[1][n * IN_DIM + k] = (_Float16)Wu[t];
    }
    __syncthreads();

    const int wave = threadIdx.x >> 5;
    const int lane = threadIdx.x & 31;
    const int half = lane >> 4;
    const int r    = lane & 15;

    int m0 = (blockIdx.x * 8 + wave) * 16;
    if (m0 > M - 16) m0 = M - 16;          // clamp: duplicate work, same values

    // ---- A fragments (16x32 f16, K=0..31 and K=32..63) ----
    // lane(r,half) holds row m0+r, K = 16h..16h+15 (+32 for a1)
    const long long uid = uids[m0 + r];
    const float* xr = emb + uid * (long long)IN_DIM;
    v16h a0, a1;
    #pragma unroll
    for (int t = 0; t < 8; ++t) {
        a0[t]     = (_Float16)xr[16 * half + t];
        a0[8 + t] = (_Float16)xr[16 * half + 8 + t];
        a1[t]     = (_Float16)xr[32 + 16 * half + t];
        a1[8 + t] = (_Float16)xr[40 + 16 * half + t];
    }

    #pragma unroll
    for (int br = 0; br < 2; ++br) {
        _Float16* hsel = br ? hu : hb;
        #pragma unroll
        for (int nb = 0; nb < 2; ++nb) {
            // B fragment: lane(r,half) holds col n=16*nb+r, K=16*half..16*half+15
            // transposed LDS layout makes this one contiguous 32-byte read.
            const int nbase = (16 * nb + r) * IN_DIM + 16 * half;
            v16h b0 = *reinterpret_cast<const v16h*>(&sW[br][nbase]);
            v16h b1 = *reinterpret_cast<const v16h*>(&sW[br][nbase + 32]);

            v8f c = {};
            c = __builtin_amdgcn_wmma_f32_16x16x32_f16(false, a0, false, b0,
                                                       (short)0, c, false, false);
            c = __builtin_amdgcn_wmma_f32_16x16x32_f16(false, a1, false, b1,
                                                       (short)0, c, false, false);
            // D layout: VGPR i -> row i + 8*half, col r of this 16x16 tile
            #pragma unroll
            for (int i = 0; i < 8; ++i)
                sT[wave][(i + 8 * half) * OUT_DIM + 16 * nb + r] = (_Float16)c[i];
        }
        __syncthreads();
        // write tile rows contiguously: lane L -> row L/2, cols (L&1)*16..+15
        {
            int ri = lane >> 1;
            int ch = (lane & 1) * 16;
            const uint4* src = reinterpret_cast<const uint4*>(&sT[wave][ri * OUT_DIM + ch]);
            uint4* dst = reinterpret_cast<uint4*>(hsel + (long long)(m0 + ri) * OUT_DIM + ch);
            dst[0] = src[0];
            dst[1] = src[1];
        }
        __syncthreads();
    }
}

// ---------------- source-side scalar projection: p[i] = h[i] . a[:32] ----------------
__global__ void k_proj(const _Float16* __restrict__ h,
                       const float* __restrict__ a,
                       float* __restrict__ p, int B) {
    int i = blockIdx.x * blockDim.x + threadIdx.x;
    if (i >= B) return;
    const _Float16* hr = h + (long long)i * OUT_DIM;
    float s = 0.f;
    #pragma unroll
    for (int c = 0; c < OUT_DIM; ++c) s += (float)hr[c] * a[c];
    p[i] = s;
}

// ---------------- edges: score, exp, atomic segment-sum ----------------
// 8 lanes per edge; each lane owns 4 of the 32 output channels.
__global__ __launch_bounds__(256)
void k_edges(const _Float16* __restrict__ h,
             const float* __restrict__ a_dst,      // a + 32
             const float* __restrict__ p_src,
             const int* __restrict__ self_cols,
             const int* __restrict__ e_rows,
             const int* __restrict__ e_cols,
             float* __restrict__ num,
             float* __restrict__ denom,
             int B, int E) {
    long long gt   = (long long)blockIdx.x * blockDim.x + threadIdx.x;
    long long edge = gt >> 3;
    int l8 = (int)(gt & 7);
    if (edge >= (long long)B + E) return;

    int row, col;
    if (edge < B) { row = (int)edge; col = self_cols[edge]; }
    else          { row = e_rows[edge - B]; col = e_cols[edge - B]; }

    v4h hh = *reinterpret_cast<const v4h*>(h + (long long)col * OUT_DIM + l8 * 4);
    float hv[4], part = 0.f;
    #pragma unroll
    for (int t = 0; t < 4; ++t) { hv[t] = (float)hh[t]; part += hv[t] * a_dst[l8 * 4 + t]; }
    part += __shfl_xor(part, 1, 8);
    part += __shfl_xor(part, 2, 8);
    part += __shfl_xor(part, 4, 8);

    float s  = p_src[row] + part;
    float sc = s > 0.f ? s : ALPHA * s;       // leaky_relu
    float w  = __expf(-sc);

    if (l8 == 0) atomic_add_f32(&denom[row], w);
    float* nr = num + (long long)row * OUT_DIM + l8 * 4;
    #pragma unroll
    for (int t = 0; t < 4; ++t) atomic_add_f32(&nr[t], w * hv[t]);
}

// ---------------- finalize: elu(num/denom) ----------------
__global__ void k_finish(const float* __restrict__ num,
                         const float* __restrict__ denom,
                         float* __restrict__ out, int B) {
    int i = blockIdx.x * blockDim.x + threadIdx.x;
    if (i >= B * OUT_DIM) return;
    float v = num[i] / denom[i >> 5];
    out[i] = v > 0.f ? v : (__expf(v) - 1.f);
}

// ---------------- launch ----------------
extern "C" void kernel_launch(void* const* d_in, const int* in_sizes, int n_in,
                              void* d_out, int out_size, void* d_ws, size_t ws_size,
                              hipStream_t stream) {
    const float* emb       = (const float*)d_in[0];
    const float* W_bal     = (const float*)d_in[1];
    const float* a_bal     = (const float*)d_in[2];
    const float* W_unbal   = (const float*)d_in[3];
    const float* a_unbal   = (const float*)d_in[4];
    const int*   uids      = (const int*)d_in[5];
    const int*   self_cols = (const int*)d_in[6];
    const int*   pos_rows  = (const int*)d_in[7];
    const int*   pos_cols  = (const int*)d_in[8];
    const int*   neg_rows  = (const int*)d_in[9];
    const int*   neg_cols  = (const int*)d_in[10];

    const int M = in_sizes[5];
    const int B = in_sizes[6];
    const int E = in_sizes[7];

    // workspace layout (256B aligned)
    char* ws = (char*)d_ws;
    size_t off = 0;
    auto alloc = [&](size_t bytes) { char* p = ws + off; off = (off + bytes + 255) & ~(size_t)255; return p; };
    _Float16* hb   = (_Float16*)alloc((size_t)M * OUT_DIM * sizeof(_Float16));
    _Float16* hu   = (_Float16*)alloc((size_t)M * OUT_DIM * sizeof(_Float16));
    float*    psb  = (float*)alloc((size_t)B * sizeof(float));
    float*    psu  = (float*)alloc((size_t)B * sizeof(float));
    // keep accumulators contiguous so one zero pass covers them
    float*    numb = (float*)alloc((size_t)B * (2 * OUT_DIM + 2) * sizeof(float));
    float*    numu = (float*)(numb + (size_t)B * OUT_DIM);
    float*    denb = (float*)(numu + (size_t)B * OUT_DIM);
    float*    denu = (float*)(denb + B);

    float* out_bal   = (float*)d_out;
    float* out_unbal = out_bal + (size_t)B * OUT_DIM;

    // 1) zero num/denom accumulators
    {
        long long n = (long long)B * (2 * OUT_DIM + 2);
        int blocks = (int)((n + 255) / 256);
        k_zero<<<blocks, 256, 0, stream>>>(numb, n);
    }
    // 2) gather + dual GEMM via WMMA
    {
        int tiles  = (M + 15) / 16;
        int blocks = (tiles + 7) / 8;
        k_gemm<<<blocks, 256, 0, stream>>>(emb, uids, W_bal, W_unbal, hb, hu, M);
    }
    // 3) source projections (first B rows only)
    {
        int blocks = (B + 255) / 256;
        k_proj<<<blocks, 256, 0, stream>>>(hb, a_bal, psb, B);
        k_proj<<<blocks, 256, 0, stream>>>(hu, a_unbal, psu, B);
    }
    // 4) edge attention + segment sums
    {
        long long threads = ((long long)B + E) * 8;
        int blocks = (int)((threads + 255) / 256);
        k_edges<<<blocks, 256, 0, stream>>>(hb, a_bal + OUT_DIM, psb,
                                            self_cols, pos_rows, pos_cols,
                                            numb, denb, B, E);
        k_edges<<<blocks, 256, 0, stream>>>(hu, a_unbal + OUT_DIM, psu,
                                            self_cols, neg_rows, neg_cols,
                                            numu, denu, B, E);
    }
    // 5) elu(num/denom)
    {
        int blocks = (B * OUT_DIM + 255) / 256;
        k_finish<<<blocks, 256, 0, stream>>>(numb, denb, out_bal, B);
        k_finish<<<blocks, 256, 0, stream>>>(numu, denu, out_unbal, B);
    }
}
